// FeedForwardCustom_63196148793884
// MI455X (gfx1250) — compile-verified, matrix operand order checked by reading, not run on hardware
//
#include <hip/hip_runtime.h>
#include <math.h>

// ---------------------------------------------------------------------------
// Problem: B=4,S=2048,D=2048,H=8192 ; M=B*S=8192
//   GEMM1/3: [M,D]x[H,D]^T -> [M,H] (K=2048), fused SiLU-gate epilogue
//   FWHT over H per row (1/sqrt(H))
//   GEMM2  : [M,H]x[D,H]^T -> [M,D] (K=8192)
// ~0.83 TFLOP vs ~320MB traffic -> compute-bound -> v_wmma_f32_16x16x32_bf16.
// Staging uses CDNA5 async-to-LDS (ASYNCcnt) with double-buffered tiles.
// ---------------------------------------------------------------------------

#define MTOK  8192
#define DDIM  2048
#define HDIM  8192

typedef __attribute__((ext_vector_type(4)))  unsigned int u32x4;
typedef __attribute__((ext_vector_type(16))) __bf16       bf16x16;
typedef __attribute__((ext_vector_type(8)))  float        f32x8;

union Frag { u32x4 u[2]; bf16x16 v; };

__device__ __forceinline__ unsigned short f2bf(float f) {
  union { float f; unsigned int u; } x; x.f = f;
  unsigned int r = x.u + 0x7FFFu + ((x.u >> 16) & 1u);   // RNE
  return (unsigned short)(r >> 16);
}
__device__ __forceinline__ float bf2f(unsigned short b) {
  union { float f; unsigned int u; } x; x.u = ((unsigned int)b) << 16;
  return x.f;
}

// CDNA5 async global->LDS copy (bypasses VGPRs, tracked by ASYNCcnt).
// LDS address = low 32 bits of the flat shared-aperture pointer (ISA 10.2).
__device__ __forceinline__ void async_ld_b128(void* lds, const void* gptr) {
  unsigned laddr = (unsigned)(unsigned long long)lds;
  asm volatile("global_load_async_to_lds_b128 %0, %1, off"
               :: "v"(laddr), "v"(gptr) : "memory");
}
#define WAIT_ASYNC(n) asm volatile("s_wait_asynccnt %0" :: "n"(n) : "memory")

// ---------------------------------------------------------------------------
// Kernel 0: fp32 -> bf16 convert
// ---------------------------------------------------------------------------
__global__ void cvt_f32_bf16(const float* __restrict__ in,
                             unsigned short* __restrict__ out, int n) {
  int i = blockIdx.x * blockDim.x + threadIdx.x;
  int stride = gridDim.x * blockDim.x;
  for (; i < n; i += stride) out[i] = f2bf(in[i]);
}

// ---------------------------------------------------------------------------
// Tiling: 256 thr = 8 wave32 (4 M x 2 N); block tile 128x64x64;
// wave tile 32x32 = 2x2 WMMA tiles. LDS rows padded to 80 elems (160B).
// ---------------------------------------------------------------------------
#define BM  128
#define BN  64
#define BK  64
#define LDT 80

__device__ __forceinline__ bf16x16 ld_frag(const unsigned short* tile,
                                           int row0, int kk, int lane) {
  const int g = lane >> 4;
  const int r = lane & 15;
  const unsigned short* p = tile + (row0 + r) * LDT + kk + (g << 3);
  Frag f;
  f.u[0] = *(const u32x4*)(p);
  f.u[1] = *(const u32x4*)(p + 16);
  return f.v;
}

// ---------------------------------------------------------------------------
// Kernel 1: hidden = silu(x@W1^T) * (x@W3^T)  (bf16 -> bf16)
// Per-wave async instructions per tile: A 4 + B1 2 + B3 2 = 8
// ---------------------------------------------------------------------------
__global__ __launch_bounds__(256)
void gemm_swiglu_wmma(const unsigned short* __restrict__ Ax,
                      const unsigned short* __restrict__ Bw1,
                      const unsigned short* __restrict__ Bw3,
                      unsigned short* __restrict__ Hid,
                      int M, int N, int K) {
  __shared__ unsigned short As[2][BM * LDT];
  __shared__ unsigned short B1s[2][BN * LDT];
  __shared__ unsigned short B3s[2][BN * LDT];

  const int tid  = threadIdx.x;
  const int lane = tid & 31;
  const int w    = tid >> 5;
  const int wr   = w & 3;
  const int wc   = w >> 2;
  const int m0   = blockIdx.y * BM;
  const int n0   = blockIdx.x * BN;

  f32x8 acc1[2][2], acc3[2][2];
#pragma unroll
  for (int i = 0; i < 2; ++i)
#pragma unroll
    for (int j = 0; j < 2; ++j) {
      acc1[i][j] = (f32x8){0.f,0.f,0.f,0.f,0.f,0.f,0.f,0.f};
      acc3[i][j] = (f32x8){0.f,0.f,0.f,0.f,0.f,0.f,0.f,0.f};
    }

  const int ktiles = K / BK;

  // Stage one K-tile (async, 8 instructions per wave)
  auto stage = [&](int k0, int b) {
#pragma unroll
    for (int c = 0; c < 4; ++c) {
      int ch = tid + c * 256, rr = ch >> 3, cc = (ch & 7) << 3;
      async_ld_b128(&As[b][rr * LDT + cc],
                    &Ax[(size_t)(m0 + rr) * K + k0 + cc]);
    }
#pragma unroll
    for (int c = 0; c < 2; ++c) {
      int ch = tid + c * 256, rr = ch >> 3, cc = (ch & 7) << 3;
      async_ld_b128(&B1s[b][rr * LDT + cc],
                    &Bw1[(size_t)(n0 + rr) * K + k0 + cc]);
      async_ld_b128(&B3s[b][rr * LDT + cc],
                    &Bw3[(size_t)(n0 + rr) * K + k0 + cc]);
    }
  };

  stage(0, 0);

  for (int kt = 0; kt < ktiles; ++kt) {
    const int buf = kt & 1;
    if (kt + 1 < ktiles) {
      stage((kt + 1) * BK, buf ^ 1);       // prefill other buffer (async)
      // L2 prefetch two tiles ahead
      if (kt + 2 < ktiles)
        __builtin_prefetch(&Ax[(size_t)(m0 + (tid >> 1)) * K +
                               (kt + 2) * BK + ((tid & 1) << 5)], 0, 1);
      WAIT_ASYNC(8);                       // current tile's 8 loads retired
    } else {
      WAIT_ASYNC(0);
    }
    __syncthreads();

#pragma unroll
    for (int kk = 0; kk < BK; kk += 32) {
      bf16x16 a[2], b1[2], b3[2];
#pragma unroll
      for (int i = 0; i < 2; ++i)
        a[i] = ld_frag(As[buf], wr * 32 + i * 16, kk, lane);
#pragma unroll
      for (int j = 0; j < 2; ++j) {
        b1[j] = ld_frag(B1s[buf], wc * 32 + j * 16, kk, lane);
        b3[j] = ld_frag(B3s[buf], wc * 32 + j * 16, kk, lane);
      }
#pragma unroll
      for (int i = 0; i < 2; ++i)
#pragma unroll
        for (int j = 0; j < 2; ++j) {
          acc1[i][j] = __builtin_amdgcn_wmma_f32_16x16x32_bf16(
              false, a[i], false, b1[j], (short)0, acc1[i][j], false, false);
          acc3[i][j] = __builtin_amdgcn_wmma_f32_16x16x32_bf16(
              false, a[i], false, b3[j], (short)0, acc3[i][j], false, false);
        }
    }
    __syncthreads();   // all waves done reading buf before it is re-staged
  }

  // Epilogue: silu(gate)*lin -> bf16 (C/D layout: col=l&15, row=r+8*(l>>4))
#pragma unroll
  for (int i = 0; i < 2; ++i)
#pragma unroll
    for (int j = 0; j < 2; ++j) {
      const int gm = m0 + wr * 32 + i * 16 + ((lane >> 4) << 3);
      const int gn = n0 + wc * 32 + j * 16 + (lane & 15);
#pragma unroll
      for (int r = 0; r < 8; ++r) {
        float g = acc1[i][j][r];
        float l = acc3[i][j][r];
        float s = g / (1.0f + __expf(-g));
        Hid[(size_t)(gm + r) * N + gn] = f2bf(s * l);
      }
    }
}

// ---------------------------------------------------------------------------
// Kernel 2: in-place FWHT along H per row, scaled 1/sqrt(H). 32KB LDS fp32.
// ---------------------------------------------------------------------------
__global__ __launch_bounds__(256)
void fwht_rows(unsigned short* __restrict__ hid, int Hn, float scale) {
  __shared__ float s[HDIM];
  const int tid = threadIdx.x;
  unsigned short* p = hid + (size_t)blockIdx.x * Hn;

  for (int i = tid; i < Hn; i += 256) s[i] = bf2f(p[i]);
  __syncthreads();

  for (int h = 1; h < Hn; h <<= 1) {
    for (int t = tid; t < (Hn >> 1); t += 256) {
      int ia = ((t & ~(h - 1)) << 1) | (t & (h - 1));
      int ib = ia + h;
      float a = s[ia], b = s[ib];
      s[ia] = a + b;
      s[ib] = a - b;
    }
    __syncthreads();
  }

  for (int i = tid; i < Hn; i += 256) p[i] = f2bf(s[i] * scale);
}

// ---------------------------------------------------------------------------
// Kernel 3: out = hidden @ W2^T (fp32 out).
// Per-wave async instructions per tile: A 4 + B 2 = 6
// ---------------------------------------------------------------------------
__global__ __launch_bounds__(256)
void gemm_out_wmma(const unsigned short* __restrict__ Ax,
                   const unsigned short* __restrict__ Bw,
                   float* __restrict__ Out,
                   int M, int N, int K) {
  __shared__ unsigned short As[2][BM * LDT];
  __shared__ unsigned short Bs[2][BN * LDT];

  const int tid  = threadIdx.x;
  const int lane = tid & 31;
  const int w    = tid >> 5;
  const int wr   = w & 3;
  const int wc   = w >> 2;
  const int m0   = blockIdx.y * BM;
  const int n0   = blockIdx.x * BN;

  f32x8 acc[2][2];
#pragma unroll
  for (int i = 0; i < 2; ++i)
#pragma unroll
    for (int j = 0; j < 2; ++j)
      acc[i][j] = (f32x8){0.f,0.f,0.f,0.f,0.f,0.f,0.f,0.f};

  const int ktiles = K / BK;

  auto stage = [&](int k0, int b) {
#pragma unroll
    for (int c = 0; c < 4; ++c) {
      int ch = tid + c * 256, rr = ch >> 3, cc = (ch & 7) << 3;
      async_ld_b128(&As[b][rr * LDT + cc],
                    &Ax[(size_t)(m0 + rr) * K + k0 + cc]);
    }
#pragma unroll
    for (int c = 0; c < 2; ++c) {
      int ch = tid + c * 256, rr = ch >> 3, cc = (ch & 7) << 3;
      async_ld_b128(&Bs[b][rr * LDT + cc],
                    &Bw[(size_t)(n0 + rr) * K + k0 + cc]);
    }
  };

  stage(0, 0);

  for (int kt = 0; kt < ktiles; ++kt) {
    const int buf = kt & 1;
    if (kt + 1 < ktiles) {
      stage((kt + 1) * BK, buf ^ 1);
      if (kt + 2 < ktiles)
        __builtin_prefetch(&Ax[(size_t)(m0 + (tid >> 1)) * K +
                               (kt + 2) * BK + ((tid & 1) << 5)], 0, 1);
      WAIT_ASYNC(6);
    } else {
      WAIT_ASYNC(0);
    }
    __syncthreads();

#pragma unroll
    for (int kk = 0; kk < BK; kk += 32) {
      bf16x16 a[2], b[2];
#pragma unroll
      for (int i = 0; i < 2; ++i)
        a[i] = ld_frag(As[buf], wr * 32 + i * 16, kk, lane);
#pragma unroll
      for (int j = 0; j < 2; ++j)
        b[j] = ld_frag(Bs[buf], wc * 32 + j * 16, kk, lane);
#pragma unroll
      for (int i = 0; i < 2; ++i)
#pragma unroll
        for (int j = 0; j < 2; ++j)
          acc[i][j] = __builtin_amdgcn_wmma_f32_16x16x32_bf16(
              false, a[i], false, b[j], (short)0, acc[i][j], false, false);
    }
    __syncthreads();
  }

#pragma unroll
  for (int i = 0; i < 2; ++i)
#pragma unroll
    for (int j = 0; j < 2; ++j) {
      const int gm = m0 + wr * 32 + i * 16 + ((lane >> 4) << 3);
      const int gn = n0 + wc * 32 + j * 16 + (lane & 15);
#pragma unroll
      for (int r = 0; r < 8; ++r)
        Out[(size_t)(gm + r) * N + gn] = acc[i][j][r];
    }
}

// ---------------------------------------------------------------------------
// Host-side launcher
// ---------------------------------------------------------------------------
extern "C" void kernel_launch(void* const* d_in, const int* in_sizes, int n_in,
                              void* d_out, int out_size, void* d_ws, size_t ws_size,
                              hipStream_t stream) {
  const float* x  = (const float*)d_in[0];   // [M, D]
  const float* W1 = (const float*)d_in[1];   // [H, D]
  const float* W2 = (const float*)d_in[2];   // [D, H]
  const float* W3 = (const float*)d_in[3];   // [H, D]
  float* out = (float*)d_out;                // [M, D]

  const int M = MTOK, D = DDIM, H = HDIM;

  char* ws = (char*)d_ws;
  unsigned short* xb  = (unsigned short*)(ws);
  unsigned short* w1b = (unsigned short*)(ws + (size_t)M * D * 2);
  unsigned short* w3b = (unsigned short*)(ws + (size_t)M * D * 2 + (size_t)H * D * 2);
  unsigned short* w2b = (unsigned short*)(ws + (size_t)M * D * 2 + (size_t)H * D * 4);
  unsigned short* hid = (unsigned short*)(ws + (size_t)M * D * 2 + (size_t)H * D * 6);

  {
    int n = M * D;
    cvt_f32_bf16<<<4096, 256, 0, stream>>>(x, xb, n);
    n = H * D;
    cvt_f32_bf16<<<4096, 256, 0, stream>>>(W1, w1b, n);
    cvt_f32_bf16<<<4096, 256, 0, stream>>>(W3, w3b, n);
    n = D * H;
    cvt_f32_bf16<<<4096, 256, 0, stream>>>(W2, w2b, n);
  }
  {
    dim3 grid(H / BN, M / BM);
    gemm_swiglu_wmma<<<grid, 256, 0, stream>>>(xb, w1b, w3b, hid, M, H, D);
  }
  {
    const float scale = 1.0f / sqrtf((float)H);
    fwht_rows<<<M, 256, 0, stream>>>(hid, H, scale);
  }
  {
    dim3 grid(D / BN, M / BM);
    gemm_out_wmma<<<grid, 256, 0, stream>>>(hid, w2b, out, M, D, H);
  }
}
